// WeightedGCNBlock_41686952575093
// MI455X (gfx1250) — compile-verified, matrix operand order checked by reading, not run on hardware
//
#include <hip/hip_runtime.h>
#include <hip/hip_bf16.h>
#include <stdint.h>

#define N_NODES 100000
#define FDIM    128
#define N_EDGES 1600000
#define BN_EPS  1e-5f

typedef __attribute__((ext_vector_type(16))) __bf16 v16bf;
typedef __attribute__((ext_vector_type(8)))  float  v8f;

union Frag16 { v16bf v; uint32_t u[8]; };
union AccU   { v8f   v; float    f[8]; };

__device__ __forceinline__ uint16_t f32_to_bf16(float x) {
    union { float f; uint32_t u; } c; c.f = x;
    uint32_t r = c.u + 0x7FFFu + ((c.u >> 16) & 1u);  // round-to-nearest-even
    return (uint16_t)(r >> 16);
}

__global__ void k_zero(float* p, int n) {
    int i = blockIdx.x * blockDim.x + threadIdx.x;
    if (i < n) p[i] = 0.0f;
}

__global__ void k_f32_to_bf16(const float* __restrict__ in, uint16_t* __restrict__ out, int n) {
    int i = blockIdx.x * blockDim.x + threadIdx.x;
    if (i < n) out[i] = f32_to_bf16(in[i]);
}

// deg[d] += w[e] over all edges (self-loop +1 added in k_dinv)
__global__ void k_deg(const int* __restrict__ dst, const float* __restrict__ w,
                      float* __restrict__ deg) {
    int e = blockIdx.x * blockDim.x + threadIdx.x;
    if (e < N_EDGES) unsafeAtomicAdd(&deg[dst[e]], w[e]);
}

__global__ void k_dinv(const float* __restrict__ deg, float* __restrict__ dinv) {
    int i = blockIdx.x * blockDim.x + threadIdx.x;
    if (i < N_NODES) dinv[i] = rsqrtf(deg[i] + 1.0f);
}

__global__ void k_norm(const int* __restrict__ src, const int* __restrict__ dst,
                       const float* __restrict__ w, const float* __restrict__ dinv,
                       float* __restrict__ nrm) {
    int e = blockIdx.x * blockDim.x + threadIdx.x;
    if (e < N_EDGES) nrm[e] = dinv[src[e]] * w[e] * dinv[dst[e]];
}

// x = h @ W^T via v_wmma_f32_16x16x32_bf16; also writes out_init = x*dinv^2 + b.
// Block = 256 threads = 8 waves; each wave computes a 32-row x 128-col strip
// (two 16-row accumulator banks share every B fragment -> halves LDS traffic).
// W is staged into LDS with GLOBAL_LOAD_ASYNC_TO_LDS_B128 (ASYNCcnt-tracked).
__global__ __launch_bounds__(256) void k_gemm_wmma(
    const uint16_t* __restrict__ hbf,   // [N,128] bf16
    const uint16_t* __restrict__ Wbf,   // [128,128] bf16, row-major W[j][k]
    const float*    __restrict__ dinv,  // [N]
    const float*    __restrict__ bias,  // [128]
    float*          __restrict__ x,     // [N,128]
    float*          __restrict__ outi)  // [N,128]
{
    __shared__ uint16_t Wlds[FDIM * FDIM];  // 32 KB
    {   // async per-lane 128-bit copies global -> LDS, drained via ASYNCcnt
        uint32_t lds0 = (uint32_t)(uintptr_t)(&Wlds[0]);
        uint64_t g0   = (uint64_t)(uintptr_t)Wbf;
        for (int i = threadIdx.x; i < (FDIM * FDIM) / 8; i += 256) {
            uint32_t la = lds0 + (uint32_t)i * 16u;
            uint64_t ga = g0 + (uint64_t)i * 16u;
            asm volatile("global_load_async_to_lds_b128 %0, %1, off"
                         :: "v"(la), "v"(ga) : "memory");
        }
        asm volatile("s_wait_asynccnt 0" ::: "memory");
    }
    __syncthreads();

    const int wave = threadIdx.x >> 5;
    const int lane = threadIdx.x & 31;
    const int half = lane >> 4;       // K half-split per ISA 16-bit operand layout
    const int l16  = lane & 15;       // A: row M; B: col N
    const int rowbase = blockIdx.x * 256 + wave * 32;

    // Per-lane K offsets (pairs are contiguous -> dword loads), ISA 7.12.2
    int koff[8];
#pragma unroll
    for (int v = 0; v < 8; ++v)
        koff[v] = (v < 4) ? (half * 8 + 2 * v) : (16 + half * 8 + 2 * (v - 4));

    AccU acc0[8], acc1[8];
#pragma unroll
    for (int nt = 0; nt < 8; ++nt)
#pragma unroll
        for (int j = 0; j < 8; ++j) { acc0[nt].f[j] = 0.0f; acc1[nt].f[j] = 0.0f; }

    const int arow0 = min(rowbase + l16,      N_NODES - 1);
    const int arow1 = min(rowbase + 16 + l16, N_NODES - 1);
    const uint16_t* hrow0 = hbf + (size_t)arow0 * FDIM;
    const uint16_t* hrow1 = hbf + (size_t)arow1 * FDIM;

#pragma unroll
    for (int kt = 0; kt < 4; ++kt) {          // K = 128 in steps of 32
        Frag16 a0, a1;
#pragma unroll
        for (int v = 0; v < 8; ++v) {
            a0.u[v] = *(const uint32_t*)(hrow0 + kt * 32 + koff[v]);
            a1.u[v] = *(const uint32_t*)(hrow1 + kt * 32 + koff[v]);
        }
#pragma unroll
        for (int nt = 0; nt < 8; ++nt) {      // 8 column tiles of 16
            Frag16 bf;
            const uint16_t* wrow = Wlds + (nt * 16 + l16) * FDIM + kt * 32;
#pragma unroll
            for (int v = 0; v < 8; ++v)
                bf.u[v] = *(const uint32_t*)(wrow + koff[v]);
            acc0[nt].v = __builtin_amdgcn_wmma_f32_16x16x32_bf16(
                false, a0.v, false, bf.v, (short)0, acc0[nt].v, false, false);
            acc1[nt].v = __builtin_amdgcn_wmma_f32_16x16x32_bf16(
                false, a1.v, false, bf.v, (short)0, acc1[nt].v, false, false);
        }
    }

    // Epilogue: C/D layout -> row = rowbase + g*16 + r + 8*half, col = nt*16 + l16
    float d2[2][8];
#pragma unroll
    for (int g = 0; g < 2; ++g)
#pragma unroll
        for (int r = 0; r < 8; ++r) {
            int row = rowbase + g * 16 + r + 8 * half;
            float dv = (row < N_NODES) ? dinv[row] : 0.0f;
            d2[g][r] = dv * dv;
        }
#pragma unroll
    for (int nt = 0; nt < 8; ++nt) {
        int col = nt * 16 + l16;
        float bc = bias[col];
#pragma unroll
        for (int g = 0; g < 2; ++g) {
            AccU* acc = (g == 0) ? acc0 : acc1;
#pragma unroll
            for (int r = 0; r < 8; ++r) {
                int row = rowbase + g * 16 + r + 8 * half;
                if (row < N_NODES) {
                    size_t idx = (size_t)row * FDIM + col;
                    float xv = acc[nt].f[r];
                    x[idx]    = xv;
                    outi[idx] = xv * d2[g][r] + bc;
                }
            }
        }
    }
}

// One wave per edge: float4 gather of x[src], scale by norm, fp32 atomic scatter to out[dst].
__global__ __launch_bounds__(256) void k_scatter(
    const int* __restrict__ src, const int* __restrict__ dst,
    const float* __restrict__ nrm, const float* __restrict__ x,
    float* __restrict__ out)
{
    int e = blockIdx.x * 8 + (threadIdx.x >> 5);
    int lane = threadIdx.x & 31;
    if (e >= N_EDGES) return;
    int s = src[e], d = dst[e];
    float n = nrm[e];
    const float4 v = *(const float4*)(x + (size_t)s * FDIM + lane * 4);
    float* o = out + (size_t)d * FDIM + lane * 4;
    unsafeAtomicAdd(o + 0, v.x * n);
    unsafeAtomicAdd(o + 1, v.y * n);
    unsafeAtomicAdd(o + 2, v.z * n);
    unsafeAtomicAdd(o + 3, v.w * n);
}

// Per-feature sum / sumsq partials over a 512-row slab, merged via fp32 atomics.
__global__ __launch_bounds__(256) void k_bn_stats(const float* __restrict__ out,
                                                  float* __restrict__ acc /* [256] */) {
    __shared__ float ssum[256], ssq[256];
    int col = threadIdx.x & 127;
    int ro  = threadIdx.x >> 7;      // 0..1
    int r0  = blockIdx.x * 512;
    int rend = min(r0 + 512, N_NODES);
    float s = 0.0f, q = 0.0f;
    for (int r = r0 + ro; r < rend; r += 2) {
        float v = out[(size_t)r * FDIM + col];
        s += v; q += v * v;
    }
    ssum[threadIdx.x] = s; ssq[threadIdx.x] = q;
    __syncthreads();
    if (threadIdx.x < 128) {
        s = ssum[threadIdx.x] + ssum[threadIdx.x + 128];
        q = ssq[threadIdx.x]  + ssq[threadIdx.x + 128];
        unsafeAtomicAdd(&acc[col], s);
        unsafeAtomicAdd(&acc[128 + col], q);
    }
}

// Fused BatchNorm(train, biased var) + ReLU; also emits next layer's bf16 activations.
__global__ __launch_bounds__(256) void k_bn_apply(
    const float* __restrict__ out, const float* __restrict__ acc,
    const float* __restrict__ gamma, const float* __restrict__ beta,
    float* __restrict__ hnext, uint16_t* __restrict__ hbf)
{
    size_t i4 = ((size_t)blockIdx.x * 256 + threadIdx.x) * 4;
    if (i4 >= (size_t)N_NODES * FDIM) return;
    int col = (int)(i4 & 127);
    const float invN = 1.0f / (float)N_NODES;
    float4 v = *(const float4*)(out + i4);
    float r[4] = {v.x, v.y, v.z, v.w};
    uint16_t bf[4];
#pragma unroll
    for (int j = 0; j < 4; ++j) {
        int c = col + j;
        float mu  = acc[c] * invN;
        float var = acc[128 + c] * invN - mu * mu;
        float y = (r[j] - mu) * rsqrtf(var + BN_EPS) * gamma[c] + beta[c];
        y = fmaxf(y, 0.0f);
        r[j] = y;
        bf[j] = f32_to_bf16(y);
    }
    float4 w4 = {r[0], r[1], r[2], r[3]};
    *(float4*)(hnext + i4) = w4;
    if (hbf) {
        uint2 p;
        p.x = (uint32_t)bf[0] | ((uint32_t)bf[1] << 16);
        p.y = (uint32_t)bf[2] | ((uint32_t)bf[3] << 16);
        *(uint2*)(hbf + i4) = p;
    }
}

extern "C" void kernel_launch(void* const* d_in, const int* in_sizes, int n_in,
                              void* d_out, int out_size, void* d_ws, size_t ws_size,
                              hipStream_t stream) {
    (void)in_sizes; (void)n_in; (void)out_size; (void)ws_size;
    const float* h0  = (const float*)d_in[0];
    const int*   ei  = (const int*)d_in[1];
    const int*   src = ei;
    const int*   dst = ei + N_EDGES;
    const float* w   = (const float*)d_in[2];
    const float* Wm[3] = {(const float*)d_in[3], (const float*)d_in[7],  (const float*)d_in[11]};
    const float* bv[3] = {(const float*)d_in[4], (const float*)d_in[8],  (const float*)d_in[12]};
    const float* gm[3] = {(const float*)d_in[5], (const float*)d_in[9],  (const float*)d_in[13]};
    const float* bt[3] = {(const float*)d_in[6], (const float*)d_in[10], (const float*)d_in[14]};

    char* ws = (char*)d_ws;
    size_t off = 0;
    auto alloc = [&](size_t bytes) -> void* {
        void* p = ws + off;
        off = (off + bytes + 255) & ~(size_t)255;
        return p;
    };
    float*    bufA  = (float*)alloc((size_t)N_NODES * FDIM * 4);   // x / next-h
    float*    bufB  = (float*)alloc((size_t)N_NODES * FDIM * 4);   // out accumulator
    uint16_t* hbf   = (uint16_t*)alloc((size_t)N_NODES * FDIM * 2);
    float*    normb = (float*)alloc((size_t)N_EDGES * 4);
    float*    degb  = (float*)alloc((size_t)N_NODES * 4);
    float*    dinvb = (float*)alloc((size_t)N_NODES * 4);
    uint16_t* Wbf   = (uint16_t*)alloc((size_t)FDIM * FDIM * 2);
    float*    bnacc = (float*)alloc(2 * FDIM * 4);

    // Edge-structure precompute (shared by all 3 layers)
    k_zero<<<(N_NODES + 255) / 256, 256, 0, stream>>>(degb, N_NODES);
    k_deg<<<(N_EDGES + 255) / 256, 256, 0, stream>>>(dst, w, degb);
    k_dinv<<<(N_NODES + 255) / 256, 256, 0, stream>>>(degb, dinvb);
    k_norm<<<(N_EDGES + 255) / 256, 256, 0, stream>>>(src, dst, w, dinvb, normb);
    k_f32_to_bf16<<<(N_NODES * FDIM + 255) / 256, 256, 0, stream>>>(h0, hbf, N_NODES * FDIM);

    for (int l = 0; l < 3; ++l) {
        k_f32_to_bf16<<<(FDIM * FDIM + 255) / 256, 256, 0, stream>>>(Wm[l], Wbf, FDIM * FDIM);
        k_gemm_wmma<<<(N_NODES + 255) / 256, 256, 0, stream>>>(hbf, Wbf, dinvb, bv[l], bufA, bufB);
        k_scatter<<<N_EDGES / 8, 256, 0, stream>>>(src, dst, normb, bufA, bufB);
        k_zero<<<1, 256, 0, stream>>>(bnacc, 2 * FDIM);
        k_bn_stats<<<(N_NODES + 511) / 512, 256, 0, stream>>>(bufB, bnacc);
        float*    dbuf = (l == 2) ? (float*)d_out : bufA;
        uint16_t* bdst = (l == 2) ? nullptr : hbf;
        k_bn_apply<<<(N_NODES * FDIM / 4 + 255) / 256, 256, 0, stream>>>(
            bufB, bnacc, gm[l], bt[l], dbuf, bdst);
    }
}
